// VirTuesEncoder_77197742179052
// MI455X (gfx1250) — matrix operand order repaired and checked by reading, not compiled, not run on hardware
//
#include <hip/hip_runtime.h>
#include <hip/hip_bf16.h>
#include <math.h>

// ---------------------------------------------------------------------------
// Types for CDNA5 WMMA (wave32): D = A(16x32 bf16) * B(32x16 bf16) + C(f32)
// ---------------------------------------------------------------------------
typedef unsigned short ushort_t;
typedef __bf16 bf16_t;
typedef bf16_t v16bf __attribute__((ext_vector_type(16)));
typedef float v8f __attribute__((ext_vector_type(8)));
typedef int v4i __attribute__((ext_vector_type(4)));

union BFrag { v16bf v; unsigned int u[8]; ushort_t h[16]; };
union FAcc { v8f v; float f[8]; };

#define Bb 4
#define C0DIM 16
#define Cc 17
#define Ss 256
#define DM 512
#define FF 1024
#define NHEADS 8
#define TOK (Bb * Cc * Ss) /* 17408 */

// Async global->LDS DMA (CDNA5). Guarded: host pass / older toolchains fall
// back to the synchronous uint4 staging path. Builtin signature (from hipcc
// diagnostics): (global int4*, local int4*, imm offset, imm cpol).
#if defined(__HIP_DEVICE_COMPILE__) && \
    __has_builtin(__builtin_amdgcn_global_load_async_to_lds_b128)
#define GEMM_ASYNC 1
typedef __attribute__((address_space(1))) v4i* as1_i4ptr;
typedef __attribute__((address_space(3))) v4i* as3_i4ptr;
__device__ __forceinline__ void async_wait_all() {
#if __has_builtin(__builtin_amdgcn_s_wait_asynccnt)
  __builtin_amdgcn_s_wait_asynccnt(0);
#else
  asm volatile("s_wait_asynccnt 0" ::: "memory");
#endif
}
#else
#define GEMM_ASYNC 0
#endif

__device__ __forceinline__ ushort_t f2bf(float x) {
  union { float f; unsigned int u; } c; c.f = x;
  unsigned int u = c.u;
  return (ushort_t)((u + 0x7FFFu + ((u >> 16) & 1u)) >> 16);
}

__device__ __forceinline__ float gelu_tanh(float x) {
  float x3 = x * x * x;
  return 0.5f * x * (1.0f + tanhf(0.7978845608028654f * (x + 0.044715f * x3)));
}

// ---------------------------------------------------------------------------
// f32 -> bf16 weight conversion (grid-stride)
// ---------------------------------------------------------------------------
__global__ __launch_bounds__(256) void f2bf_kernel(const float* __restrict__ in,
                                                   ushort_t* __restrict__ out, int n) {
  int i = blockIdx.x * blockDim.x + threadIdx.x;
  int stride = gridDim.x * blockDim.x;
  for (; i < n; i += stride) out[i] = f2bf(in[i]);
}

// ---------------------------------------------------------------------------
// WMMA GEMM: C[M,N] = A[M,K](f32, cvt->bf16 in LDS) * Bw[K,N](bf16) + bias
// Block tile 64x256, BK=32, double-buffered LDS. 8 waves each own a 32x64
// tile (2x4 accumulators -> 8 WMMA per wave per K-step from 2 A-frag + 4
// B-frag LDS reads). B tiles staged via async global->LDS DMA when available.
// M%64==0, N%256==0, K%32==0 hold for every call site here.
// ---------------------------------------------------------------------------
__global__ __launch_bounds__(256) void wmma_gemm_bf16(
    const float* __restrict__ A, const ushort_t* __restrict__ Bw,
    const float* __restrict__ bias, float* __restrict__ C,
    int M, int N, int K, int applyGelu) {
  __shared__ __align__(16) ushort_t As[2][64][40];
  __shared__ __align__(16) ushort_t Bs[2][32][264];
  const int tid = threadIdx.x;
  const int lane = tid & 31;
  const int wid = tid >> 5;
  const int bm = blockIdx.y * 64;
  const int bn = blockIdx.x * 256;
  const int wm = (wid >> 2) * 32;  // 2 wave rows
  const int wn = (wid & 3) * 64;   // 4 wave cols x 64 wide

  FAcc acc[2][4];
#pragma unroll
  for (int i = 0; i < 2; ++i)
#pragma unroll
    for (int j = 0; j < 4; ++j)
#pragma unroll
      for (int r = 0; r < 8; ++r) acc[i][j].f[r] = 0.f;

  const int arow = tid >> 2;        // 0..63
  const int acol = (tid & 3) * 8;   // 0,8,16,24
  const int brow = tid >> 3;        // 0..31
  const int bcol = (tid & 7) * 32;  // 0..224 (32 halves = 4x b128 per thread)

  auto stageA = [&](int buf, int k0) {
    const float* src = A + (size_t)(bm + arow) * K + (k0 + acol);
    float4 f0 = *(const float4*)src;
    float4 f1 = *(const float4*)(src + 4);
    ushort_t* d = &As[buf][arow][acol];
    d[0] = f2bf(f0.x); d[1] = f2bf(f0.y); d[2] = f2bf(f0.z); d[3] = f2bf(f0.w);
    d[4] = f2bf(f1.x); d[5] = f2bf(f1.y); d[6] = f2bf(f1.z); d[7] = f2bf(f1.w);
    if (k0 + 32 < K) __builtin_prefetch(src + 32, 0, 1);  // global_prefetch_b8
  };
  auto stageB = [&](int buf, int k0) {
    const ushort_t* src = Bw + (size_t)(k0 + brow) * N + (bn + bcol);
    ushort_t* dst = &Bs[buf][brow][bcol];
#if GEMM_ASYNC
#pragma unroll
    for (int i = 0; i < 4; ++i)
      __builtin_amdgcn_global_load_async_to_lds_b128(
          (as1_i4ptr)(src + i * 8), (as3_i4ptr)(dst + i * 8), 0, 0);
#else
    const uint4* s4 = (const uint4*)src;
    uint4* d4 = (uint4*)dst;
#pragma unroll
    for (int i = 0; i < 4; ++i) d4[i] = s4[i];
#endif
  };

  stageA(0, 0);
  stageB(0, 0);
#if GEMM_ASYNC
  async_wait_all();
#endif
  __syncthreads();

  const int mlo = lane & 15;
  const int khi = (lane >> 4) * 8;  // lanes 16-31 hold K 8..15 / 24..31
  int buf = 0;
  for (int k0 = 0; k0 < K; k0 += 32) {
    const int nxt = buf ^ 1;
    const bool more = (k0 + 32 < K);
    if (more) {  // overlap next-tile staging with this tile's WMMAs
      stageA(nxt, k0 + 32);
      stageB(nxt, k0 + 32);
    }

    BFrag afr[2], bfr[4];
#pragma unroll
    for (int ms = 0; ms < 2; ++ms) {
      const int m = wm + ms * 16 + mlo;
#pragma unroll
      for (int vv = 0; vv < 8; ++vv) {
        const int kk = ((vv < 4) ? 0 : 16) + khi + (vv & 3) * 2;
        afr[ms].u[vv] = *(const unsigned int*)&As[buf][m][kk];
      }
    }
#pragma unroll
    for (int ns = 0; ns < 4; ++ns) {
#pragma unroll
      for (int s = 0; s < 16; ++s)
        bfr[ns].h[s] = Bs[buf][((s < 8) ? 0 : 16) + khi + (s & 7)][wn + ns * 16 + mlo];
    }
#pragma unroll
    for (int ms = 0; ms < 2; ++ms)
#pragma unroll
      for (int ns = 0; ns < 4; ++ns)
        acc[ms][ns].v = __builtin_amdgcn_wmma_f32_16x16x32_bf16(
            false, afr[ms].v, false, bfr[ns].v, (short)0, acc[ms][ns].v, false, false);

    if (more) {
#if GEMM_ASYNC
      async_wait_all();
#endif
      __syncthreads();
    }
    buf = nxt;
  }

  const int nl = lane & 15;
  const int mo = (lane >> 4) * 8;
#pragma unroll
  for (int ms = 0; ms < 2; ++ms) {
#pragma unroll
    for (int ns = 0; ns < 4; ++ns) {
      const int gn = bn + wn + ns * 16 + nl;
      const float bv = bias ? bias[gn] : 0.f;
#pragma unroll
      for (int r = 0; r < 8; ++r) {
        const int gm = bm + wm + ms * 16 + mo + r;
        float val = acc[ms][ns].f[r] + bv;
        if (applyGelu) val = gelu_tanh(val);
        C[(size_t)gm * N + gn] = val;
      }
    }
  }
}

// ---------------------------------------------------------------------------
// Attention: one workgroup = (16-query tile, head, sequence). Whole K/V head
// lives in LDS (320KB WGP budget). QK^T and P.V both via bf16 WMMA; softmax
// in fp32 with wave32 shuffles; partial P.V combined via ds_add_f32 atomics.
// ---------------------------------------------------------------------------
__global__ __launch_bounds__(256) void attn_wmma_kernel(
    const float* __restrict__ q, const float* __restrict__ k,
    const float* __restrict__ v, float* __restrict__ o, int L, int Lp) {
  extern __shared__ __align__(16) char smem[];
  const int tid = threadIdx.x;
  const int lane = tid & 31;
  const int wid = tid >> 5;
  const int qt = blockIdx.x;
  const int h = blockIdx.y;
  const int n = blockIdx.z;
  const int SKT = Lp + 8;

  ushort_t* sQ = (ushort_t*)smem;                 // 16 x 72 (bf16)
  ushort_t* sKt = sQ + 16 * 72;                   // 64 x SKT (K transposed)
  ushort_t* sV = sKt + 64 * SKT;                  // Lp x 72
  float* sS = (float*)(sV + (size_t)Lp * 72);     // 16 x SKT (scores f32)
  ushort_t* sP = (ushort_t*)(sS + 16 * SKT);      // 16 x SKT (probs bf16)
  float* sO = (float*)(sP + 16 * SKT);            // 16 x 72

  const size_t tokBase = ((size_t)n * L) * DM + h * 64;

  for (int idx = tid; idx < 16 * 64; idx += 256) {
    int r = idx >> 6, d = idx & 63;
    int row = qt * 16 + r;
    float f = (row < L) ? q[tokBase + (size_t)row * DM + d] : 0.f;
    sQ[r * 72 + d] = f2bf(f);
  }
  for (int idx = tid; idx < Lp * 64; idx += 256) {
    int r = idx >> 6, d = idx & 63;
    float fk = 0.f, fv = 0.f;
    if (r < L) {
      fk = k[tokBase + (size_t)r * DM + d];
      fv = v[tokBase + (size_t)r * DM + d];
    }
    sKt[d * SKT + r] = f2bf(fk);
    sV[r * 72 + d] = f2bf(fv);
  }
  for (int idx = tid; idx < 16 * 72; idx += 256) sO[idx] = 0.f;
  __syncthreads();

  const int mlo = lane & 15;
  const int khi = (lane >> 4) * 8;
  const int nTiles = Lp >> 4;
  for (int j = wid; j < nTiles; j += 8) {  // scores tile (16q x 16key)
    FAcc acc;
#pragma unroll
    for (int r = 0; r < 8; ++r) acc.f[r] = 0.f;
#pragma unroll
    for (int d0 = 0; d0 < 64; d0 += 32) {
      BFrag a, b;
#pragma unroll
      for (int vv = 0; vv < 8; ++vv) {
        const int kk = d0 + ((vv < 4) ? 0 : 16) + khi + (vv & 3) * 2;
        a.u[vv] = *(const unsigned int*)&sQ[mlo * 72 + kk];
      }
#pragma unroll
      for (int s = 0; s < 16; ++s)
        b.h[s] = sKt[(d0 + ((s < 8) ? 0 : 16) + khi + (s & 7)) * SKT + j * 16 + mlo];
      acc.v = __builtin_amdgcn_wmma_f32_16x16x32_bf16(false, a.v, false, b.v,
                                                      (short)0, acc.v, false, false);
    }
    const int mo = (lane >> 4) * 8;
#pragma unroll
    for (int r = 0; r < 8; ++r)
      sS[(mo + r) * SKT + j * 16 + mlo] = acc.f[r] * 0.125f;  // 1/sqrt(64)
  }
  __syncthreads();

  { // softmax: 16 threads per row, wave32 shuffles within 16-lane groups
    const int row = tid >> 4;
    const int t = tid & 15;
    float mx = -1e30f;
    for (int c = t; c < L; c += 16) mx = fmaxf(mx, sS[row * SKT + c]);
#pragma unroll
    for (int i = 1; i < 16; i <<= 1) mx = fmaxf(mx, __shfl_xor(mx, i, 16));
    float sum = 0.f;
    for (int c = t; c < Lp; c += 16) {
      float e = (c < L) ? __expf(sS[row * SKT + c] - mx) : 0.f;
      sS[row * SKT + c] = e;
      sum += e;
    }
#pragma unroll
    for (int i = 1; i < 16; i <<= 1) sum += __shfl_xor(sum, i, 16);
    const float inv = 1.f / sum;
    for (int c = t; c < Lp; c += 16) sP[row * SKT + c] = f2bf(sS[row * SKT + c] * inv);
  }
  __syncthreads();

  { // O = P(16xLp) . V(Lpx64): waves split (d-tile, K-half); combine in LDS
    const int nt = wid & 3;
    const int kh = wid >> 2;
    const int kSteps = Lp >> 5;
    FAcc acc;
#pragma unroll
    for (int r = 0; r < 8; ++r) acc.f[r] = 0.f;
    bool any = false;
    for (int s = kh; s < kSteps; s += 2) {
      const int kb = s * 32;
      BFrag a, b;
#pragma unroll
      for (int vv = 0; vv < 8; ++vv) {
        const int kk = kb + ((vv < 4) ? 0 : 16) + khi + (vv & 3) * 2;
        a.u[vv] = *(const unsigned int*)&sP[mlo * SKT + kk];
      }
#pragma unroll
      for (int s = 0; s < 16; ++s)
        b.h[s] = sV[(kb + ((s < 8) ? 0 : 16) + khi + (s & 7)) * 72 + nt * 16 + mlo];
      acc.v = __builtin_amdgcn_wmma_f32_16x16x32_bf16(false, a.v, false, b.v,
                                                      (short)0, acc.v, false, false);
      any = true;
    }
    if (any) {
      const int mo = (lane >> 4) * 8;
#pragma unroll
      for (int r = 0; r < 8; ++r)
        atomicAdd(&sO[(mo + r) * 72 + nt * 16 + mlo], acc.f[r]);  // ds_add_f32
    }
  }
  __syncthreads();

  for (int idx = tid; idx < 16 * 64; idx += 256) {
    int r = idx >> 6, d = idx & 63;
    int row = qt * 16 + r;
    if (row < L) o[tokBase + (size_t)row * DM + d] = sO[r * 72 + d];
  }
}

// ---------------------------------------------------------------------------
// 2D RoPE. mode 0 ('h'): s = token pos in sequence; mode 1 ('v'): s from
// the (b*S+s) sequence index (all channel tokens share the spatial angle).
// ---------------------------------------------------------------------------
__global__ __launch_bounds__(256) void rope_kernel(float* __restrict__ x, int L, int mode) {
  const int tok = blockIdx.x;
  const int s = mode ? ((tok / L) & 255) : (tok % L);
  const float px = (float)(s >> 4);
  const float py = (float)(s & 15);
  const int p = threadIdx.x;
  const int head = p >> 5;
  const int within = p & 31;
  const int half = within >> 4;
  const int i = within & 15;
  const float freq = __powf(10000.0f, -(float)i * (1.0f / 16.0f));
  const float theta = (half ? py : px) * freq;
  const float c = __cosf(theta), sn = __sinf(theta);
  float* basep = x + (size_t)tok * DM + head * 64 + half * 32 + 2 * i;
  const float v1 = basep[0], v2 = basep[1];
  basep[0] = v1 * c - v2 * sn;
  basep[1] = v1 * sn + v2 * c;
}

// ---------------------------------------------------------------------------
// x = LayerNorm(x + y) * g + b   (one block per 512-wide token row)
// ---------------------------------------------------------------------------
__global__ __launch_bounds__(256) void add_ln_kernel(
    float* __restrict__ x, const float* __restrict__ y,
    const float* __restrict__ g, const float* __restrict__ b) {
  __shared__ float red[8];
  __shared__ float stat;
  const int t = threadIdx.x;
  const int lane = t & 31, wid = t >> 5;
  const size_t base = (size_t)blockIdx.x * DM;
  const float v0 = x[base + t] + y[base + t];
  const float v1 = x[base + t + 256] + y[base + t + 256];
  float sacc = v0 + v1;
#pragma unroll
  for (int i = 16; i >= 1; i >>= 1) sacc += __shfl_xor(sacc, i, 32);
  if (lane == 0) red[wid] = sacc;
  __syncthreads();
  if (t == 0) {
    float tot = 0.f;
    for (int i = 0; i < 8; ++i) tot += red[i];
    stat = tot * (1.f / 512.f);
  }
  __syncthreads();
  const float mu = stat;
  const float d0 = v0 - mu, d1 = v1 - mu;
  float qv = d0 * d0 + d1 * d1;
#pragma unroll
  for (int i = 16; i >= 1; i >>= 1) qv += __shfl_xor(qv, i, 32);
  if (lane == 0) red[wid] = qv;
  __syncthreads();
  if (t == 0) {
    float tot = 0.f;
    for (int i = 0; i < 8; ++i) tot += red[i];
    stat = rsqrtf(tot * (1.f / 512.f) + 1e-5f);
  }
  __syncthreads();
  const float rs = stat;
  x[base + t] = d0 * rs * g[t] + b[t];
  x[base + t + 256] = d1 * rs * g[t + 256] + b[t + 256];
}

// ---------------------------------------------------------------------------
// prot[row] = protein_emb[ids[row]] @ prot_enc_W + bias    (64 rows, K=1280)
// ---------------------------------------------------------------------------
__global__ __launch_bounds__(256) void prot_gemm_kernel(
    const float* __restrict__ emb, const int* __restrict__ ids,
    const float* __restrict__ W, const float* __restrict__ bias,
    float* __restrict__ out) {
  const int row = blockIdx.x;
  const int id = ids[row];
  const float* e = emb + (size_t)id * 1280;
  const int ncol = threadIdx.x;
  float a0 = bias[ncol], a1 = bias[ncol + 256];
  for (int kk = 0; kk < 1280; ++kk) {
    const float ev = e[kk];
    a0 = fmaf(ev, W[(size_t)kk * DM + ncol], a0);
    a1 = fmaf(ev, W[(size_t)kk * DM + ncol + 256], a1);
  }
  out[(size_t)row * DM + ncol] = a0;
  out[(size_t)row * DM + ncol + 256] = a1;
}

// ---------------------------------------------------------------------------
// Assemble x(B,17,S,512): channel 0 = summary token; 1..16 = patch + prot
// ---------------------------------------------------------------------------
__global__ __launch_bounds__(256) void assemble_kernel(
    const float* __restrict__ t, const float* __restrict__ prot,
    const float* __restrict__ summary, float* __restrict__ xb) {
  const int tok = blockIdx.x;
  const int b = tok / (Cc * Ss);
  const int rem = tok % (Cc * Ss);
  const int c = rem >> 8;
  const int s = rem & 255;
  float* dst = xb + (size_t)tok * DM;
  const int d0 = threadIdx.x, d1 = threadIdx.x + 256;
  if (c == 0) {
    dst[d0] = summary[d0];
    dst[d1] = summary[d1];
  } else {
    const size_t srow = ((size_t)((b * C0DIM + (c - 1)) * Ss + s)) * DM;
    const size_t prow = (size_t)(b * C0DIM + (c - 1)) * DM;
    dst[d0] = t[srow + d0] + prot[prow + d0];
    dst[d1] = t[srow + d1] + prot[prow + d1];
  }
}

// in (B,D1,D2,512) -> out (B,D2,D1,512)
__global__ __launch_bounds__(128) void transpose_kernel(
    const float* __restrict__ in, float* __restrict__ out, int D1, int D2) {
  const int tok = blockIdx.x;
  const int b = tok / (D1 * D2);
  const int r = tok % (D1 * D2);
  const int i = r / D2;
  const int j = r % D2;
  const float4* src = (const float4*)(in + (size_t)tok * DM);
  float4* dst = (float4*)(out + ((size_t)((b * D2 + j) * D1 + i)) * DM);
  dst[threadIdx.x] = src[threadIdx.x];
}

// out = concat( x[:,1:] flat , x[:,0] flat )
__global__ __launch_bounds__(128) void write_out_kernel(const float* __restrict__ xb,
                                                        float* __restrict__ out) {
  const int tok = blockIdx.x;
  const int b = tok / (Cc * Ss);
  const int rem = tok % (Cc * Ss);
  const int c = rem >> 8;
  const int s = rem & 255;
  const float4* src = (const float4*)(xb + (size_t)tok * DM);
  float4* dst;
  if (c == 0)
    dst = (float4*)(out + (size_t)8388608 + ((size_t)(b * Ss + s)) * DM);
  else
    dst = (float4*)(out + ((size_t)((b * C0DIM + (c - 1)) * Ss + s)) * DM);
  dst[threadIdx.x] = src[threadIdx.x];
}

// ---------------------------------------------------------------------------
// Host orchestration
// ---------------------------------------------------------------------------
extern "C" void kernel_launch(void* const* d_in, const int* in_sizes, int n_in,
                              void* d_out, int out_size, void* d_ws, size_t ws_size,
                              hipStream_t stream) {
  const float* x_in = (const float*)d_in[0];
  const int* channel_ids = (const int*)d_in[1];
  const float* protein_emb = (const float*)d_in[2];
  const float* summary = (const float*)d_in[3];
  const float* patchW = (const float*)d_in[4];
  const float* patchB = (const float*)d_in[5];
  const float* protW = (const float*)d_in[6];
  const float* protBias = (const float*)d_in[7];

  struct LP {
    const float *Wq, *Wk, *Wv, *Wo, *bq, *bk, *bv, *bo;
    const float *g1, *b1, *W1, *c1, *W2, *c2, *g2, *b2;
  } lp[4];
  // JAX pytree flattens dicts by sorted key; detect via first layer entry size.
  const bool sortedKeys = (n_in > 8) && (in_sizes[8] == 524288);
  for (int li = 0; li < 4; ++li) {
    const float* p[16];
    for (int i = 0; i < 16; ++i) p[i] = (const float*)d_in[8 + li * 16 + i];
    if (sortedKeys) {
      // [W1,W2,Wk,Wo,Wq,Wv,b1,b2,bk,bo,bq,bv,ln1_b,ln1_g,ln2_b,ln2_g]
      lp[li] = {p[4], p[2], p[5], p[3], p[10], p[8], p[11], p[9],
                p[13], p[12], p[0], p[6], p[1], p[7], p[15], p[14]};
    } else {
      // insertion order: Wq,Wk,Wv,Wo,bq,bk,bv,bo,ln1_g,ln1_b,W1,b1,W2,b2,ln2_g,ln2_b
      lp[li] = {p[0], p[1], p[2], p[3], p[4], p[5], p[6], p[7],
                p[8], p[9], p[10], p[11], p[12], p[13], p[14], p[15]};
    }
  }

  // Workspace carve (aliased where lifetimes allow)
  char* ws = (char*)d_ws;
  size_t off = 0;
  auto carve = [&](size_t bytes) -> void* {
    void* pp = ws + off;
    off = (off + bytes + 255) & ~(size_t)255;
    return pp;
  };
  ushort_t* bPatch = (ushort_t*)carve((size_t)576 * 512 * 2);
  ushort_t *bWq[4], *bWk[4], *bWv[4], *bWo[4], *bW1[4], *bW2[4];
  for (int li = 0; li < 4; ++li) {
    bWq[li] = (ushort_t*)carve((size_t)512 * 512 * 2);
    bWk[li] = (ushort_t*)carve((size_t)512 * 512 * 2);
    bWv[li] = (ushort_t*)carve((size_t)512 * 512 * 2);
    bWo[li] = (ushort_t*)carve((size_t)512 * 512 * 2);
    bW1[li] = (ushort_t*)carve((size_t)512 * 1024 * 2);
    bW2[li] = (ushort_t*)carve((size_t)1024 * 512 * 2);
  }
  float* xbuf = (float*)carve((size_t)TOK * DM * 4);
  float* x2buf = (float*)carve((size_t)TOK * DM * 4);
  float* big = (float*)carve((size_t)TOK * FF * 4);  // q | k ; reused as h
  float* qb = big;
  float* kb = big + (size_t)TOK * DM;
  float* hb = big;
  float* vb = (float*)carve((size_t)TOK * DM * 4);   // also o2 and patch-temp
  float* ob = (float*)carve((size_t)TOK * DM * 4);
  float* protb = (float*)carve((size_t)64 * DM * 4);
  float* o2b = vb;
  float* tpatch = vb;

  // bf16 weight conversion (idempotent, deterministic)
  auto conv = [&](const float* s, ushort_t* dp, int nElem) {
    int blocks = (nElem + 1023) / 1024;
    f2bf_kernel<<<blocks, 256, 0, stream>>>(s, dp, nElem);
  };
  conv(patchW, bPatch, 576 * 512);
  for (int li = 0; li < 4; ++li) {
    conv(lp[li].Wq, bWq[li], 512 * 512);
    conv(lp[li].Wk, bWk[li], 512 * 512);
    conv(lp[li].Wv, bWv[li], 512 * 512);
    conv(lp[li].Wo, bWo[li], 512 * 512);
    conv(lp[li].W1, bW1[li], 512 * 1024);
    conv(lp[li].W2, bW2[li], 1024 * 512);
  }

  // Embedding
  prot_gemm_kernel<<<64, 256, 0, stream>>>(protein_emb, channel_ids, protW, protBias, protb);
  {
    dim3 g(512 / 256, 16384 / 64);
    wmma_gemm_bf16<<<g, 256, 0, stream>>>(x_in, bPatch, patchB, tpatch, 16384, 512, 576, 0);
  }
  assemble_kernel<<<TOK, 256, 0, stream>>>(tpatch, protb, summary, xbuf);

  (void)hipFuncSetAttribute(reinterpret_cast<const void*>(attn_wmma_kernel),
                            hipFuncAttributeMaxDynamicSharedMemorySize, 160 * 1024);

  // Encoder layers, pattern 'hvhv'
  for (int li = 0; li < 4; ++li) {
    const bool isV = (li & 1);
    const int Nb = isV ? Bb * Ss : Bb * Cc;
    const int L = isV ? Cc : Ss;
    const int Lp = isV ? 32 : 256;
    float* act = isV ? x2buf : xbuf;
    if (isV) transpose_kernel<<<TOK, 128, 0, stream>>>(xbuf, x2buf, Cc, Ss);

    dim3 gp(512 / 256, TOK / 64);
    wmma_gemm_bf16<<<gp, 256, 0, stream>>>(act, bWq[li], lp[li].bq, qb, TOK, 512, 512, 0);
    wmma_gemm_bf16<<<gp, 256, 0, stream>>>(act, bWk[li], lp[li].bk, kb, TOK, 512, 512, 0);
    wmma_gemm_bf16<<<gp, 256, 0, stream>>>(act, bWv[li], lp[li].bv, vb, TOK, 512, 512, 0);
    rope_kernel<<<TOK, 256, 0, stream>>>(qb, L, isV ? 1 : 0);
    rope_kernel<<<TOK, 256, 0, stream>>>(kb, L, isV ? 1 : 0);

    const size_t smem = (size_t)(16 * 72 + 64 * (Lp + 8) + (size_t)Lp * 72) * 2 +
                        (size_t)16 * (Lp + 8) * 4 + (size_t)16 * (Lp + 8) * 2 +
                        (size_t)16 * 72 * 4;
    dim3 ga((L + 15) / 16, NHEADS, Nb);
    attn_wmma_kernel<<<ga, 256, smem, stream>>>(qb, kb, vb, ob, L, Lp);

    wmma_gemm_bf16<<<gp, 256, 0, stream>>>(ob, bWo[li], lp[li].bo, o2b, TOK, 512, 512, 0);
    add_ln_kernel<<<TOK, 256, 0, stream>>>(act, o2b, lp[li].g1, lp[li].b1);

    dim3 gf(1024 / 256, TOK / 64);
    wmma_gemm_bf16<<<gf, 256, 0, stream>>>(act, bW1[li], lp[li].c1, hb, TOK, 1024, 512, 1);
    wmma_gemm_bf16<<<gp, 256, 0, stream>>>(hb, bW2[li], lp[li].c2, o2b, TOK, 512, 1024, 0);
    add_ln_kernel<<<TOK, 256, 0, stream>>>(act, o2b, lp[li].g2, lp[li].b2);

    if (isV) transpose_kernel<<<TOK, 128, 0, stream>>>(x2buf, xbuf, Ss, Cc);
  }

  write_out_kernel<<<TOK, 128, 0, stream>>>(xbuf, (float*)d_out);
  (void)in_sizes; (void)n_in; (void)out_size; (void)ws_size;
}
